// NAttention_79568564126295
// MI455X (gfx1250) — matrix operand order repaired and checked by reading, not compiled, not run on hardware
//
#include <hip/hip_runtime.h>

typedef __attribute__((ext_vector_type(16))) __bf16 v16bf;
typedef __attribute__((ext_vector_type(8)))  __bf16 v8bf;
typedef __attribute__((ext_vector_type(8)))  float  v8f;

#define NN   2
#define TT   4096
#define CC   256
#define HH   4
#define HD   64
#define WINW 64
#define LEFTW 31
#define SCALEF 0.125f
#define MASKV  (-100.0f)
#define TPAD 4208          // 31 left pad + 4096 + 81 right pad, multiple of 16
#define VCOLS 144          // staged V window per 64-query block (multiple of 16)

static __device__ __forceinline__ __bf16 f2bf(float f) { return (__bf16)f; }
static __device__ __forceinline__ int iclampi(int v, int lo, int hi) {
  return v < lo ? lo : (v > hi ? hi : v);
}

// gfx1250 async global->LDS copy (ASYNCcnt path, CDNA5 ISA).
// ldsOff = low 32 bits of the flat shared address (aperture rule, ISA 10.2).
static __device__ __forceinline__ void async_copy_b128(unsigned ldsOff, const void* gptr) {
  asm volatile("global_load_async_to_lds_b128 %0, %1, off"
               :: "v"(ldsOff), "v"((unsigned long long)(size_t)gptr)
               : "memory");
}
static __device__ __forceinline__ void wait_async0() {
#if __has_builtin(__builtin_amdgcn_s_wait_asynccnt)
  __builtin_amdgcn_s_wait_asynccnt(0);
#else
  asm volatile("s_wait_asynccnt 0" ::: "memory");
#endif
}

// ---------------------------------------------------------------------------
// Kernel 0a: transpose fp32 weight (K x Nc, row-major) -> bf16 (Nc x K)
// ---------------------------------------------------------------------------
__global__ __launch_bounds__(256) void wt_bf16_kernel(const float* __restrict__ W,
                                                      __bf16* __restrict__ Wt,
                                                      int K, int Nc) {
  int i = blockIdx.x * blockDim.x + threadIdx.x;
  if (i >= K * Nc) return;
  int k = i % K;
  int n = i / K;
  Wt[i] = f2bf(W[(size_t)k * Nc + n]);
}

// ---------------------------------------------------------------------------
// Kernel 0b: zero the pad regions of the padded transposed V buffer.
// ---------------------------------------------------------------------------
__global__ __launch_bounds__(256) void vpad_zero_kernel(__bf16* __restrict__ vTp) {
  const int PADN = LEFTW + (TPAD - LEFTW - TT);   // 112 pad elems / row
  int i = blockIdx.x * blockDim.x + threadIdx.x;
  const int rows = NN * HH * HD;                  // 512
  if (i >= rows * PADN) return;
  int r = i / PADN, p = i % PADN;
  int t = (p < LEFTW) ? p : (LEFTW + TT + (p - LEFTW));
  vTp[(size_t)r * TPAD + t] = f2bf(0.0f);
}

// ---------------------------------------------------------------------------
// Kernel 1: qkv = x @ W_qkv + b_qkv   (M=8192, K=256, N=768)
// One wave computes a 64(M)x16(N) strip: B loaded once per K-step, reused x4.
// ---------------------------------------------------------------------------
__global__ __launch_bounds__(256) void qkv_gemm_kernel(const float* __restrict__ x,
                                                       const __bf16* __restrict__ WtQ,
                                                       const float* __restrict__ bias,
                                                       __bf16* __restrict__ qB,
                                                       __bf16* __restrict__ kB,
                                                       __bf16* __restrict__ vTp) {
  const int lane = threadIdx.x & 31;
  const int wv   = threadIdx.x >> 5;
  const int lm   = lane & 15;
  const int half = lane >> 4;
  const int mt = blockIdx.x;               // 0..127
  const int nt = blockIdx.y * 8 + wv;      // 0..47
  const int col = nt * 16 + lm;
  const __bf16* wcol = WtQ + (size_t)col * CC;

  const float bval = bias[col];
  v8f acc[4];
#pragma unroll
  for (int mi = 0; mi < 4; ++mi)
#pragma unroll
    for (int r = 0; r < 8; ++r) acc[mi][r] = bval;

#pragma unroll
  for (int kb = 0; kb < CC; kb += 32) {
    v16bf b = *(const v16bf*)(wcol + kb + half * 16);
    __builtin_prefetch(wcol + kb + 32 + half * 16, 0, 3);  // global_prefetch
#pragma unroll
    for (int mi = 0; mi < 4; ++mi) {
      const float* xrow = x + (size_t)(mt * 64 + mi * 16 + lm) * CC;
      const float4 c0 = *(const float4*)(xrow + kb + half * 16);
      const float4 c1 = *(const float4*)(xrow + kb + half * 16 + 4);
      const float4 d0 = *(const float4*)(xrow + kb + half * 16 + 8);
      const float4 d1 = *(const float4*)(xrow + kb + half * 16 + 12);
      v16bf a;
      a[0]=f2bf(c0.x); a[1]=f2bf(c0.y); a[2]=f2bf(c0.z); a[3]=f2bf(c0.w);
      a[4]=f2bf(c1.x); a[5]=f2bf(c1.y); a[6]=f2bf(c1.z); a[7]=f2bf(c1.w);
      a[8]=f2bf(d0.x); a[9]=f2bf(d0.y); a[10]=f2bf(d0.z); a[11]=f2bf(d0.w);
      a[12]=f2bf(d1.x); a[13]=f2bf(d1.y); a[14]=f2bf(d1.z); a[15]=f2bf(d1.w);
      acc[mi] = __builtin_amdgcn_wmma_f32_16x16x32_bf16(false, a, false, b, (short)0,
                                                        acc[mi], false, false);
    }
  }

  const int s  = col >> 8;
  const int h  = (col & 255) >> 6;
  const int hd = col & 63;
#pragma unroll
  for (int mi = 0; mi < 4; ++mi) {
    if (s == 0) {
#pragma unroll
      for (int r = 0; r < 8; ++r) {
        int g = mt * 64 + mi * 16 + r + half * 8;
        int n = g >> 12, t = g & (TT - 1);
        qB[(((size_t)(n * HH + h) * TT + t) * HD) + hd] = f2bf(acc[mi][r]);
      }
    } else if (s == 1) {
#pragma unroll
      for (int r = 0; r < 8; ++r) {
        int g = mt * 64 + mi * 16 + r + half * 8;
        int n = g >> 12, t = g & (TT - 1);
        kB[(((size_t)(n * HH + h) * TT + t) * HD) + hd] = f2bf(acc[mi][r]);
      }
    } else {
#pragma unroll
      for (int r = 0; r < 8; ++r) {
        int g = mt * 64 + mi * 16 + r + half * 8;
        int n = g >> 12, t = g & (TT - 1);
        vTp[(((size_t)(n * HH + h) * HD + hd) * TPAD) + LEFTW + t] = f2bf(acc[mi][r]);
      }
    }
  }
}

// ---------------------------------------------------------------------------
// Kernel 2: windowed attention. Workgroup = 4 waves = 64 queries of one (n,h).
// V tile (64 x 144 bf16) staged once per block via async global->LDS DMA;
// each wave then runs S = Q K^T (10 WMMAs), softmax, O = P V (12 WMMAs, B from LDS).
// ---------------------------------------------------------------------------
__global__ __launch_bounds__(128) void attn_win_kernel(const __bf16* __restrict__ qB,
                                                       const __bf16* __restrict__ kB,
                                                       const __bf16* __restrict__ vTp,
                                                       const float* __restrict__ rel_bias,
                                                       __bf16* __restrict__ attnO) {
  __shared__ __bf16 Vlds[HD][VCOLS];      // 64 x 144 bf16 = 18 KB
  __shared__ __bf16 Pbuf[4][16 * 96];     // 12 KB

  const int lane = threadIdx.x & 31;
  const int wv   = threadIdx.x >> 5;      // 0..3
  const int lm   = lane & 15;
  const int half = lane >> 4;

  const int b    = blockIdx.x;            // 0..511
  const int tt64 = b & 63;                // T/64
  const int nh   = b >> 6;                // 0..7
  const int h    = nh & (HH - 1);
  const int n    = nh >> 2;
  const int t0   = tt64 * 64;             // block query base
  const int t0w  = t0 + wv * 16;          // wave query base

  const __bf16* qbase = qB  + (size_t)(n * HH + h) * TT * HD;
  const __bf16* kbase = kB  + (size_t)(n * HH + h) * TT * HD;
  const __bf16* vbase = vTp + (size_t)(n * HH + h) * HD * TPAD;

  // ---- async-stage V tile: padded cols [t0, t0+144) of all 64 hd rows ----
  // 64 rows * 18 chunks of 16B = 1152 chunks = 9 per thread (no divergence).
  {
    const unsigned ldsBase = (unsigned)(size_t)(&Vlds[0][0]);
    const int tid = threadIdx.x;
#pragma unroll
    for (int it = 0; it < 9; ++it) {
      const int c   = tid + it * 128;
      const int row = c / 18;
      const int cc  = c % 18;
      const __bf16* g = vbase + (size_t)row * TPAD + t0 + cc * 8;
      const unsigned l = ldsBase + (unsigned)((row * VCOLS + cc * 8) * 2);
      async_copy_b128(l, g);
    }
    wait_async0();
    __syncthreads();
  }

  // ---- S = Q K^T over the 80-key union (5 column tiles) ----
  v8f S[5];
#pragma unroll
  for (int j = 0; j < 5; ++j)
#pragma unroll
    for (int r = 0; r < 8; ++r) S[j][r] = 0.0f;

#pragma unroll
  for (int kb = 0; kb < HD; kb += 32) {
    const __bf16* qrow = qbase + (size_t)(t0w + lm) * HD;
    v8bf qlo = *(const v8bf*)(qrow + kb + half * 16);
    v8bf qhi = *(const v8bf*)(qrow + kb + half * 16 + 8);
    v16bf a;
#pragma unroll
    for (int i = 0; i < 8; ++i) { a[i] = qlo[i]; a[8 + i] = qhi[i]; }
#pragma unroll
    for (int j = 0; j < 5; ++j) {
      int key = iclampi(t0w - LEFTW + j * 16 + lm, 0, TT - 1);   // ref clips idx too
      v16bf kv = *(const v16bf*)(kbase + (size_t)key * HD + kb + half * 16);
      S[j] = __builtin_amdgcn_wmma_f32_16x16x32_bf16(false, a, false, kv, (short)0,
                                                     S[j], false, false);
    }
  }

  // ---- scale + rel_bias + window/bounds mask (branch-free) ----
#pragma unroll
  for (int j = 0; j < 5; ++j) {
    const int jj   = j * 16 + lm;
    const int idxj = t0w - LEFTW + jj;
    const float oob = (idxj < 0 || idxj >= TT) ? MASKV : 0.0f;
#pragma unroll
    for (int r = 0; r < 8; ++r) {
      const int m = r + half * 8;
      const int w = jj - m;
      const int wi = iclampi(w, 0, WINW - 1);
      const float sc = S[j][r] * SCALEF + rel_bias[h * WINW + wi] + oob;
      S[j][r] = (w >= 0 && w < WINW) ? sc : -1.0e30f;
    }
  }

  // ---- row softmax: tile reduce + 16-lane butterfly within each half ----
#pragma unroll
  for (int r = 0; r < 8; ++r) {
    float mx = -1.0e30f;
#pragma unroll
    for (int j = 0; j < 5; ++j) mx = fmaxf(mx, S[j][r]);
    mx = fmaxf(mx, __shfl_xor(mx, 1, 32));
    mx = fmaxf(mx, __shfl_xor(mx, 2, 32));
    mx = fmaxf(mx, __shfl_xor(mx, 4, 32));
    mx = fmaxf(mx, __shfl_xor(mx, 8, 32));
    float sm = 0.0f;
#pragma unroll
    for (int j = 0; j < 5; ++j) { float e = __expf(S[j][r] - mx); S[j][r] = e; sm += e; }
    sm += __shfl_xor(sm, 1, 32);
    sm += __shfl_xor(sm, 2, 32);
    sm += __shfl_xor(sm, 4, 32);
    sm += __shfl_xor(sm, 8, 32);
    const float inv = 1.0f / sm;
#pragma unroll
    for (int j = 0; j < 5; ++j) S[j][r] *= inv;
  }

  // ---- P -> LDS (bf16, 16x96, cols 80..95 = 0) ----
  __bf16* prow = &Pbuf[wv][0];
#pragma unroll
  for (int r = 0; r < 8; ++r) {
    const int m = r + half * 8;
#pragma unroll
    for (int j = 0; j < 5; ++j) prow[m * 96 + j * 16 + lm] = f2bf(S[j][r]);
    prow[m * 96 + 80 + lm] = f2bf(0.0f);
  }

  // ---- O = P V  (K = 96 padded keys, N = HD = 64, B from staged LDS tile) ----
  v8f O[4];
#pragma unroll
  for (int j = 0; j < 4; ++j)
#pragma unroll
    for (int r = 0; r < 8; ++r) O[j][r] = 0.0f;

  const __bf16* pr = &Pbuf[wv][lm * 96];
#pragma unroll
  for (int kb = 0; kb < 96; kb += 32) {
    v8bf plo = *(const v8bf*)(pr + kb + half * 16);        // ds_load_b128
    v8bf phi = *(const v8bf*)(pr + kb + half * 16 + 8);
    v16bf a;
#pragma unroll
    for (int i = 0; i < 8; ++i) { a[i] = plo[i]; a[8 + i] = phi[i]; }
#pragma unroll
    for (int nt2 = 0; nt2 < 4; ++nt2) {
      const int hd = nt2 * 16 + lm;
      // wave-local column offset into the staged tile (32B aligned)
      v16bf vv = *(const v16bf*)(&Vlds[hd][wv * 16 + kb + half * 16]);
      O[nt2] = __builtin_amdgcn_wmma_f32_16x16x32_bf16(false, a, false, vv, (short)0,
                                                       O[nt2], false, false);
    }
  }

  // ---- store O -> attnO (n, t, h, hd) bf16 ----
#pragma unroll
  for (int nt2 = 0; nt2 < 4; ++nt2) {
    const int hd = nt2 * 16 + lm;
#pragma unroll
    for (int r = 0; r < 8; ++r) {
      const int t = t0w + r + half * 8;
      attnO[(((size_t)(n * TT + t) * HH + h) * HD) + hd] = f2bf(O[nt2][r]);
    }
  }
}

// ---------------------------------------------------------------------------
// Kernel 3: out = attnO @ W_proj + b_proj  (M=8192, K=256, N=256), f32 output.
// ---------------------------------------------------------------------------
__global__ __launch_bounds__(256) void proj_gemm_kernel(const __bf16* __restrict__ A,
                                                        const __bf16* __restrict__ WtP,
                                                        const float* __restrict__ bias,
                                                        float* __restrict__ out) {
  const int lane = threadIdx.x & 31;
  const int wv   = threadIdx.x >> 5;
  const int lm   = lane & 15;
  const int half = lane >> 4;
  const int mt = blockIdx.x;               // 0..127
  const int nt = blockIdx.y * 8 + wv;      // 0..15
  const int col = nt * 16 + lm;
  const __bf16* wcol = WtP + (size_t)col * CC;

  const float bval = bias[col];
  v8f acc[4];
#pragma unroll
  for (int mi = 0; mi < 4; ++mi)
#pragma unroll
    for (int r = 0; r < 8; ++r) acc[mi][r] = bval;

#pragma unroll
  for (int kb = 0; kb < CC; kb += 32) {
    v16bf b = *(const v16bf*)(wcol + kb + half * 16);
    __builtin_prefetch(wcol + kb + 32 + half * 16, 0, 3);
#pragma unroll
    for (int mi = 0; mi < 4; ++mi) {
      const __bf16* arow = A + (size_t)(mt * 64 + mi * 16 + lm) * CC;
      v8bf lo = *(const v8bf*)(arow + kb + half * 16);
      v8bf hi = *(const v8bf*)(arow + kb + half * 16 + 8);
      v16bf a;
#pragma unroll
      for (int i = 0; i < 8; ++i) { a[i] = lo[i]; a[8 + i] = hi[i]; }
      acc[mi] = __builtin_amdgcn_wmma_f32_16x16x32_bf16(false, a, false, b, (short)0,
                                                        acc[mi], false, false);
    }
  }

#pragma unroll
  for (int mi = 0; mi < 4; ++mi)
#pragma unroll
    for (int r = 0; r < 8; ++r) {
      const int g = mt * 64 + mi * 16 + r + half * 8;
      out[(size_t)g * CC + col] = acc[mi][r];
    }
}

// ---------------------------------------------------------------------------
extern "C" void kernel_launch(void* const* d_in, const int* in_sizes, int n_in,
                              void* d_out, int out_size, void* d_ws, size_t ws_size,
                              hipStream_t stream) {
  const float* x        = (const float*)d_in[0];   // (2,4096,256)
  const float* W_qkv    = (const float*)d_in[1];   // (256,768)
  const float* b_qkv    = (const float*)d_in[2];   // (768)
  const float* rel_bias = (const float*)d_in[3];   // (1,4,1,64)
  const float* W_proj   = (const float*)d_in[4];   // (256,256)
  const float* b_proj   = (const float*)d_in[5];   // (256)
  float* out = (float*)d_out;                      // (2,4096,256)

  size_t off = 0;
  auto carve = [&](size_t bytes) -> void* {
    void* p = (char*)d_ws + off;
    off += (bytes + 255) & ~(size_t)255;
    return p;
  };
  __bf16* WtQ = (__bf16*)carve((size_t)768 * 256 * 2);
  __bf16* WtP = (__bf16*)carve((size_t)256 * 256 * 2);
  __bf16* qB  = (__bf16*)carve((size_t)NN * HH * TT * HD * 2);
  __bf16* kB  = (__bf16*)carve((size_t)NN * HH * TT * HD * 2);
  __bf16* vTp = (__bf16*)carve((size_t)NN * HH * HD * TPAD * 2);
  __bf16* aO  = (__bf16*)carve((size_t)NN * TT * CC * 2);

  wt_bf16_kernel<<<(256 * 768 + 255) / 256, 256, 0, stream>>>(W_qkv, WtQ, 256, 768);
  wt_bf16_kernel<<<(256 * 256 + 255) / 256, 256, 0, stream>>>(W_proj, WtP, 256, 256);
  {
    const int padElems = NN * HH * HD * (LEFTW + (TPAD - LEFTW - TT));
    vpad_zero_kernel<<<(padElems + 255) / 256, 256, 0, stream>>>(vTp);
  }
  qkv_gemm_kernel<<<dim3(128, 6), 256, 0, stream>>>(x, WtQ, b_qkv, qB, kB, vTp);
  attn_win_kernel<<<dim3(512), 128, 0, stream>>>(qB, kB, vTp, rel_bias, aO);
  proj_gemm_kernel<<<dim3(128, 2), 256, 0, stream>>>(aO, WtP, b_proj, out);
}
